// Qwen3MoeDecoderLayer_41188736369345
// MI455X (gfx1250) — compile-verified, matrix operand order checked by reading, not run on hardware
//
#include <hip/hip_runtime.h>
#include <hip/hip_bf16.h>

// ---------------------------------------------------------------------------
// Qwen3-MoE decoder layer for gfx1250 (MI455X). All GEMMs use
// v_wmma_f32_16x16x32_bf16 with f32->bf16 conversion while staging into LDS.
// Attention K/V staging: double-buffered global_load_async_to_lds_b128
// (ASYNCcnt), overlapping DMA with the WMMA compute phase.
// GEMM: register-staged double buffering of global tiles across K-steps.
// ---------------------------------------------------------------------------

typedef __attribute__((ext_vector_type(16))) __bf16 v16bf;
typedef __attribute__((ext_vector_type(8)))  float  v8f;

#define T_   2048
#define H_   2048
#define NQ_  32
#define NKV_ 4
#define D_   128
#define E_   64
#define K_   8
#define I_   768
#define CAP_ 512
#define QKVN (NQ_ * D_ + 2 * NKV_ * D_)   // 5120
#define AD_  (NQ_ * D_)                   // 4096

__device__ inline v8f v8f_zero() {
  v8f z;
#pragma unroll
  for (int i = 0; i < 8; ++i) z[i] = 0.0f;
  return z;
}

__device__ inline v8f wmma_bf16(v16bf a, v16bf b, v8f c) {
  // D = A(16x32 bf16) * B(32x16 bf16) + C(16x16 f32)
  return __builtin_amdgcn_wmma_f32_16x16x32_bf16(false, a, false, b, (short)0, c,
                                                 false, false);
}

// K index inside a 32-wide K-step for A/B fragment VGPR v, given lane-half lh.
__device__ inline int frag_k(int v, int lh) {
  return (v < 4) ? (lh * 8 + 2 * v) : (16 + lh * 8 + 2 * (v - 4));
}

__device__ inline float ld_as_f32(const float*  p) { return *p; }
__device__ inline float ld_as_f32(const __bf16* p) { return (float)*p; }

// --- CDNA5 async global->LDS copy (16B per lane), tracked by ASYNCcnt. ------
__device__ inline void async_copy_b128(unsigned lds_byte_addr, const void* gaddr) {
  asm volatile("global_load_async_to_lds_b128 %0, %1, off"
               :: "v"(lds_byte_addr),
                  "v"((unsigned long long)(size_t)gaddr)
               : "memory");
}
__device__ inline void async_wait0()  { asm volatile("s_wait_asynccnt 0x0"  ::: "memory"); }
__device__ inline void async_wait16() { asm volatile("s_wait_asynccnt 0x10" ::: "memory"); }

// ---------------------------------------------------------------------------
// Generic (optionally batched) WMMA GEMM: C[M,N] = A[M,K] * B[K,N], f32 accum.
// A may be f32 or bf16 (converted to bf16 in LDS); B is f32.
// Register-staged double buffer: global loads for K-step k+1 are issued
// before the WMMA chain of K-step k, hiding HBM/L2 latency.
// MOE epilogue: row -> (token, weight) via slot tables, atomicAdd into C(=out).
// Block: 256 threads (8 waves). Block tile 128x128, K-step 32.
// ---------------------------------------------------------------------------
#define BM 128
#define BN 128
#define BK 32
#define AS_LD 34
#define BS_LD 136

template <typename TA, bool MOE>
__global__ __launch_bounds__(256, 1)
void gemm_wmma_kernel(const TA* __restrict__ A, const float* __restrict__ B,
                      float* __restrict__ C, int M, int N, int K,
                      int lda, int ldb, int ldc,
                      long long strideA, long long strideB, long long strideC,
                      const int* __restrict__ slotTok,
                      const float* __restrict__ slotW) {
  __shared__ __align__(16) __bf16 As[BM * AS_LD];
  __shared__ __align__(16) __bf16 Bs[BK * BS_LD];

  const int batch = blockIdx.z;
  A += (long long)batch * strideA;
  B += (long long)batch * strideB;
  C += (long long)batch * strideC;

  const int n0 = blockIdx.x * BN;
  const int m0 = blockIdx.y * BM;
  const int tid  = threadIdx.x;
  const int wave = tid >> 5;
  const int lane = tid & 31;
  const int lm = lane & 15;
  const int lh = lane >> 4;
  const int wm = wave & 1;   // 0..1 : 64-row half
  const int wn = wave >> 1;  // 0..3 : 32-col slice

  v8f acc[4][2];
#pragma unroll
  for (int im = 0; im < 4; ++im)
#pragma unroll
    for (int in = 0; in < 2; ++in) acc[im][in] = v8f_zero();

  float aReg[16], bReg[16];

  auto load_tile = [&](int k0) {
#pragma unroll
    for (int i = 0; i < 16; ++i) {
      int e = i * 256 + tid;
      int r = e / BK, c = e % BK;
      int gr = m0 + r;
      aReg[i] = (gr < M) ? ld_as_f32(&A[(size_t)gr * lda + (k0 + c)]) : 0.0f;
    }
#pragma unroll
    for (int i = 0; i < 16; ++i) {
      int e = i * 256 + tid;
      int r = e / BN, c = e % BN;
      int gc = n0 + c;
      bReg[i] = (gc < N) ? B[(size_t)(k0 + r) * ldb + gc] : 0.0f;
    }
  };

  load_tile(0);

  for (int k0 = 0; k0 < K; k0 += BK) {
    // --- commit staged registers to LDS as bf16 ---
#pragma unroll
    for (int i = 0; i < 16; ++i) {
      int e = i * 256 + tid;
      As[(e / BK) * AS_LD + (e % BK)] = (__bf16)aReg[i];
    }
#pragma unroll
    for (int i = 0; i < 16; ++i) {
      int e = i * 256 + tid;
      Bs[(e / BN) * BS_LD + (e % BN)] = (__bf16)bReg[i];
    }
    __syncthreads();

    // --- issue next tile's global loads (overlap with WMMA below) ---
    if (k0 + BK < K) load_tile(k0 + BK);

    // --- fragments + WMMA ---
    v16bf a[4];
#pragma unroll
    for (int im = 0; im < 4; ++im) {
      int row = wm * 64 + im * 16 + lm;
#pragma unroll
      for (int v = 0; v < 8; ++v) {
        int kb = frag_k(v, lh);
        a[im][2 * v]     = As[row * AS_LD + kb];
        a[im][2 * v + 1] = As[row * AS_LD + kb + 1];
      }
    }
    v16bf b[2];
#pragma unroll
    for (int in = 0; in < 2; ++in) {
      int col = wn * 32 + in * 16 + lm;
#pragma unroll
      for (int v = 0; v < 8; ++v) {
        int kb = frag_k(v, lh);
        b[in][2 * v]     = Bs[kb * BS_LD + col];
        b[in][2 * v + 1] = Bs[(kb + 1) * BS_LD + col];
      }
    }
#pragma unroll
    for (int im = 0; im < 4; ++im)
#pragma unroll
      for (int in = 0; in < 2; ++in)
        acc[im][in] = wmma_bf16(a[im], b[in], acc[im][in]);
    __syncthreads();
  }

  // --- epilogue ---
#pragma unroll
  for (int im = 0; im < 4; ++im) {
#pragma unroll
    for (int in = 0; in < 2; ++in) {
#pragma unroll
      for (int v = 0; v < 8; ++v) {
        int row = m0 + wm * 64 + im * 16 + v + 8 * lh;
        int col = n0 + wn * 32 + in * 16 + lm;
        if (row < M && col < N) {
          float val = acc[im][in][v];
          if constexpr (MOE) {
            int tkn = slotTok[batch * CAP_ + row];
            if (tkn >= 0) {
              atomicAdd(&C[(size_t)tkn * ldc + col], val * slotW[batch * CAP_ + row]);
            }
          } else {
            C[(size_t)row * ldc + col] = val;
          }
        }
      }
    }
  }
}

// ---------------------------------------------------------------------------
// RMSNorm over H per token. grid = T, block = 256.
// ---------------------------------------------------------------------------
__global__ void rmsnorm_kernel(const float* __restrict__ x, const float* __restrict__ w,
                               float* __restrict__ out, int Hdim) {
  __shared__ float red[256];
  const int t = blockIdx.x, tid = threadIdx.x;
  const float* xr = x + (size_t)t * Hdim;
  float acc = 0.0f;
  for (int i = tid; i < Hdim; i += 256) { float v = xr[i]; acc += v * v; }
  red[tid] = acc; __syncthreads();
  for (int off = 128; off > 0; off >>= 1) {
    if (tid < off) red[tid] += red[tid + off];
    __syncthreads();
  }
  float inv = rsqrtf(red[0] / (float)Hdim + 1e-6f);
  float* orow = out + (size_t)t * Hdim;
  for (int i = tid; i < Hdim; i += 256) orow[i] = xr[i] * inv * w[i];
}

// ---------------------------------------------------------------------------
// Per-head RMSNorm + RoPE on q/k, plus v pass-through; emits bf16 head-major
// tensors Qh[NQ][T][D], Kh[NKV][T][D], Vh[NKV][T][D]. grid = T, block = 128.
// ---------------------------------------------------------------------------
__global__ __launch_bounds__(128)
void rope_kernel(const float* __restrict__ qkv, const int* __restrict__ positions,
                 const float* __restrict__ qw, const float* __restrict__ kw,
                 __bf16* __restrict__ Qh, __bf16* __restrict__ Kh,
                 __bf16* __restrict__ Vh) {
  __shared__ float red[128];
  __shared__ float ynorm[128];
  const int t = blockIdx.x, d = threadIdx.x;
  const float pos = (float)positions[t];
  const int j = (d < 64) ? d : d - 64;
  const float ang = pos * __powf(10000.0f, -(2.0f * (float)j) / 128.0f);
  const float c = __cosf(ang), s = __sinf(ang);

  for (int hh = 0; hh < NQ_ + NKV_; ++hh) {
    const float* src = qkv + (size_t)t * QKVN + hh * D_;
    const float* w = (hh < NQ_) ? qw : kw;
    float x = src[d];
    red[d] = x * x; __syncthreads();
    for (int off = 64; off > 0; off >>= 1) {
      if (d < off) red[d] += red[d + off];
      __syncthreads();
    }
    float inv = rsqrtf(red[0] / 128.0f + 1e-6f);
    float y = x * inv * w[d];
    ynorm[d] = y; __syncthreads();
    float o = (d < 64) ? (y * c - ynorm[d + 64] * s) : (y * c + ynorm[d - 64] * s);
    if (hh < NQ_) Qh[((size_t)hh * T_ + t) * D_ + d] = (__bf16)o;
    else          Kh[((size_t)(hh - NQ_) * T_ + t) * D_ + d] = (__bf16)o;
    __syncthreads();
  }
  for (int vh = 0; vh < NKV_; ++vh) {
    float x = qkv[(size_t)t * QKVN + AD_ + NKV_ * D_ + vh * D_ + d];
    Vh[((size_t)vh * T_ + t) * D_ + d] = (__bf16)x;
  }
}

// ---------------------------------------------------------------------------
// Causal flash attention, WMMA for QK^T and PV, online softmax.
// grid = (T/64, NQ). block = 128 (4 waves, each owns 16 q rows).
// K/V tiles double-buffered in LDS via async DMA: tile kb+1 is in flight
// while tile kb is consumed by the 32-WMMA compute phase.
// ---------------------------------------------------------------------------
#define KVTILE (64 * D_)           // bf16 elements per tile
#define KVTILE_B (KVTILE * 2)      // bytes per tile

__global__ __launch_bounds__(128, 1)
void attn_kernel(const __bf16* __restrict__ Qh, const __bf16* __restrict__ Kh,
                 const __bf16* __restrict__ Vh, float* __restrict__ attn) {
  __shared__ __align__(16) __bf16 Ks[2 * KVTILE];
  __shared__ __align__(16) __bf16 Vs[2 * KVTILE];
  __shared__ __align__(16) __bf16 Ps[4][16 * 64];

  const int head = blockIdx.y;
  const int qb   = blockIdx.x * 64;
  const int wave = threadIdx.x >> 5;
  const int lane = threadIdx.x & 31;
  const int lm = lane & 15, lh = lane >> 4;
  const int kvh = head >> 3;  // NQ/NKV = 8
  const int qrow = qb + wave * 16;
  const float scale = 0.08838834764831845f;  // 1/sqrt(128)

  // Preload Q fragments (16 rows x 128 D) into registers.
  v16bf qa[4];
  const __bf16* Qbase = Qh + ((size_t)head * T_ + qrow + lm) * D_;
#pragma unroll
  for (int dc = 0; dc < 4; ++dc)
#pragma unroll
    for (int v = 0; v < 8; ++v) {
      int kb = frag_k(v, lh);
      qa[dc][2 * v]     = Qbase[dc * 32 + kb];
      qa[dc][2 * v + 1] = Qbase[dc * 32 + kb + 1];
    }

  v8f o[8];
#pragma unroll
  for (int dt = 0; dt < 8; ++dt) o[dt] = v8f_zero();
  float mrow[8], lrow[8];
#pragma unroll
  for (int v = 0; v < 8; ++v) { mrow[v] = -1e30f; lrow[v] = 0.0f; }

  const unsigned KsBase = (unsigned)(size_t)&Ks[0];
  const unsigned VsBase = (unsigned)(size_t)&Vs[0];

  // Issue one full K/V tile (16KB each = 1024 x 16B chunks -> 8 per lane each).
  auto issue_tile = [&](int kb) {
    const char* kg = (const char*)(Kh + ((size_t)kvh * T_ + kb * 64) * D_);
    const char* vg = (const char*)(Vh + ((size_t)kvh * T_ + kb * 64) * D_);
    const unsigned kbase = KsBase + (unsigned)(kb & 1) * KVTILE_B;
    const unsigned vbase = VsBase + (unsigned)(kb & 1) * KVTILE_B;
#pragma unroll
    for (int i = 0; i < 8; ++i) {
      unsigned off = (unsigned)((i * 128 + threadIdx.x) * 16);
      async_copy_b128(kbase + off, kg + off);
      async_copy_b128(vbase + off, vg + off);
    }
  };

  const int nb = blockIdx.x + 1;  // number of 64-wide kv blocks (causal)
  issue_tile(0);

  for (int kb = 0; kb < nb; ++kb) {
    // Prefetch next tile into the other buffer, then wait for current tile
    // (async loads complete in order: <=16 outstanding == current tile done).
    if (kb + 1 < nb) { issue_tile(kb + 1); async_wait16(); }
    else             { async_wait0(); }
    __syncthreads();

    const __bf16* KsT = &Ks[(kb & 1) * KVTILE];
    const __bf16* VsT = &Vs[(kb & 1) * KVTILE];
    const int kv0 = kb * 64;

    // S = Q * K^T : 4 kv subtiles x 4 D-steps.
    v8f sacc[4];
#pragma unroll
    for (int in = 0; in < 4; ++in) sacc[in] = v8f_zero();
#pragma unroll
    for (int in = 0; in < 4; ++in) {
#pragma unroll
      for (int dc = 0; dc < 4; ++dc) {
        v16bf bk;
#pragma unroll
        for (int v = 0; v < 8; ++v) {
          int kb2 = frag_k(v, lh);
          bk[2 * v]     = KsT[(in * 16 + lm) * D_ + dc * 32 + kb2];
          bk[2 * v + 1] = KsT[(in * 16 + lm) * D_ + dc * 32 + kb2 + 1];
        }
        sacc[in] = wmma_bf16(qa[dc], bk, sacc[in]);
      }
    }

    // Online softmax in C-layout (row r = v + 8*lh across a 16-lane group).
#pragma unroll
    for (int v = 0; v < 8; ++v) {
      int r = v + 8 * lh;
      int grow = qrow + r;
      float mx = -1e30f;
#pragma unroll
      for (int in = 0; in < 4; ++in) {
        int col = kv0 + in * 16 + lm;
        float xv = sacc[in][v] * scale;
        if (col > grow) xv = -1e30f;
        sacc[in][v] = xv;
        mx = fmaxf(mx, xv);
      }
      mx = fmaxf(mx, __shfl_xor(mx, 1, 16));
      mx = fmaxf(mx, __shfl_xor(mx, 2, 16));
      mx = fmaxf(mx, __shfl_xor(mx, 4, 16));
      mx = fmaxf(mx, __shfl_xor(mx, 8, 16));
      float mnew = fmaxf(mrow[v], mx);
      float alpha = __expf(mrow[v] - mnew);
      mrow[v] = mnew;
      float rs = 0.0f;
#pragma unroll
      for (int in = 0; in < 4; ++in) {
        float pv = __expf(sacc[in][v] - mnew);
        sacc[in][v] = pv;
        rs += pv;
      }
      rs += __shfl_xor(rs, 1, 16);
      rs += __shfl_xor(rs, 2, 16);
      rs += __shfl_xor(rs, 4, 16);
      rs += __shfl_xor(rs, 8, 16);
      lrow[v] = lrow[v] * alpha + rs;
#pragma unroll
      for (int dt = 0; dt < 8; ++dt) o[dt][v] *= alpha;
#pragma unroll
      for (int in = 0; in < 4; ++in)
        Ps[wave][r * 64 + in * 16 + lm] = (__bf16)sacc[in][v];
    }
    __syncthreads();

    // O += P * V : 2 K-steps x 8 D subtiles.
#pragma unroll
    for (int kc = 0; kc < 2; ++kc) {
      v16bf pa;
#pragma unroll
      for (int v = 0; v < 8; ++v) {
        int kb2 = frag_k(v, lh);
        pa[2 * v]     = Ps[wave][lm * 64 + kc * 32 + kb2];
        pa[2 * v + 1] = Ps[wave][lm * 64 + kc * 32 + kb2 + 1];
      }
#pragma unroll
      for (int dt = 0; dt < 8; ++dt) {
        v16bf bv;
#pragma unroll
        for (int v = 0; v < 8; ++v) {
          int kb2 = frag_k(v, lh);
          bv[2 * v]     = VsT[(kc * 32 + kb2) * D_ + dt * 16 + lm];
          bv[2 * v + 1] = VsT[(kc * 32 + kb2 + 1) * D_ + dt * 16 + lm];
        }
        o[dt] = wmma_bf16(pa, bv, o[dt]);
      }
    }
    __syncthreads();  // all waves done with this tile before its buffer reuse
  }

  // Normalize by l and write to attn[T][NQ*D].
#pragma unroll
  for (int v = 0; v < 8; ++v) {
    float inv = 1.0f / lrow[v];
    int grow = qrow + v + 8 * lh;
#pragma unroll
    for (int dt = 0; dt < 8; ++dt)
      attn[(size_t)grow * AD_ + head * D_ + dt * 16 + lm] = o[dt][v] * inv;
  }
}

// ---------------------------------------------------------------------------
// Elementwise helpers.
// ---------------------------------------------------------------------------
__global__ void add_kernel(const float* __restrict__ a, const float* __restrict__ b,
                           float* __restrict__ out, long long n) {
  long long i = (long long)blockIdx.x * blockDim.x + threadIdx.x;
  if (i < n) out[i] = a[i] + b[i];
}

__global__ void copy_kernel(const float* __restrict__ a, float* __restrict__ out,
                            long long n) {
  long long i = (long long)blockIdx.x * blockDim.x + threadIdx.x;
  if (i < n) out[i] = a[i];
}

__global__ void silu_mul_kernel(const float* __restrict__ g, const float* __restrict__ u,
                                __bf16* __restrict__ act, long long n) {
  long long i = (long long)blockIdx.x * blockDim.x + threadIdx.x;
  if (i < n) {
    float gv = g[i];
    float sv = gv / (1.0f + __expf(-gv));
    act[i] = (__bf16)(sv * u[i]);
  }
}

// ---------------------------------------------------------------------------
// Router: softmax over 64 experts + top-8 + renormalize. grid = T, block = 64.
// ---------------------------------------------------------------------------
__global__ void topk_kernel(const float* __restrict__ logits, float* __restrict__ topk_w,
                            int* __restrict__ topk_idx) {
  __shared__ float p[E_];
  __shared__ float rv[E_];
  __shared__ int   ri[E_];
  __shared__ float selv[K_];
  __shared__ int   seli[K_];
  const int t = blockIdx.x, e = threadIdx.x;
  float x = logits[(size_t)t * E_ + e];
  rv[e] = x; __syncthreads();
  for (int off = 32; off > 0; off >>= 1) {
    if (e < off) rv[e] = fmaxf(rv[e], rv[e + off]);
    __syncthreads();
  }
  float mx = rv[0]; __syncthreads();
  float ex = __expf(x - mx);
  rv[e] = ex; __syncthreads();
  for (int off = 32; off > 0; off >>= 1) {
    if (e < off) rv[e] += rv[e + off];
    __syncthreads();
  }
  float sum = rv[0]; __syncthreads();
  p[e] = ex / sum; __syncthreads();

  for (int k = 0; k < K_; ++k) {
    rv[e] = p[e]; ri[e] = e; __syncthreads();
    for (int off = 32; off > 0; off >>= 1) {
      if (e < off) {
        if (rv[e + off] > rv[e] || (rv[e + off] == rv[e] && ri[e + off] < ri[e])) {
          rv[e] = rv[e + off]; ri[e] = ri[e + off];
        }
      }
      __syncthreads();
    }
    if (e == 0) { selv[k] = rv[0]; seli[k] = ri[0]; p[ri[0]] = -1.0f; }
    __syncthreads();
  }
  if (e == 0) {
    float ssum = 0.0f;
    for (int k = 0; k < K_; ++k) ssum += selv[k];
    for (int k = 0; k < K_; ++k) {
      topk_w[(size_t)t * K_ + k]   = selv[k] / ssum;
      topk_idx[(size_t)t * K_ + k] = seli[k];
    }
  }
}

// ---------------------------------------------------------------------------
// Sequential capacity-based routing (matches reference cumsum ordering).
// ---------------------------------------------------------------------------
__global__ void route_kernel(const int* __restrict__ topk_idx,
                             const float* __restrict__ topk_w,
                             int* __restrict__ slot_token, float* __restrict__ slot_w) {
  if (blockIdx.x != 0 || threadIdx.x != 0) return;
  for (int s = 0; s < E_ * CAP_; ++s) { slot_token[s] = -1; slot_w[s] = 0.0f; }
  int counts[E_];
  for (int e = 0; e < E_; ++e) counts[e] = 0;
  for (int f = 0; f < T_ * K_; ++f) {
    int e = topk_idx[f];
    int pos = counts[e]++;
    if (pos < CAP_) {
      slot_token[e * CAP_ + pos] = f / K_;
      slot_w[e * CAP_ + pos]     = topk_w[f];
    }
  }
}

// Dispatch tokens into the (E, CAP, H) bf16 buffer; empty slots -> 0.
__global__ void scatter_kernel(const int* __restrict__ slot_token,
                               const float* __restrict__ h2, __bf16* __restrict__ buf) {
  const int s = blockIdx.x;
  const int t = slot_token[s];
  __bf16* row = buf + (size_t)s * H_;
  if (t >= 0) {
    const float* src = h2 + (size_t)t * H_;
    for (int c = threadIdx.x; c < H_; c += 256) row[c] = (__bf16)src[c];
  } else {
    for (int c = threadIdx.x; c < H_; c += 256) row[c] = (__bf16)0.0f;
  }
}

// ---------------------------------------------------------------------------
// Host-side orchestration.
// ---------------------------------------------------------------------------
extern "C" void kernel_launch(void* const* d_in, const int* in_sizes, int n_in,
                              void* d_out, int out_size, void* d_ws, size_t ws_size,
                              hipStream_t stream) {
  (void)in_sizes; (void)n_in; (void)out_size; (void)ws_size;
  const float* hidden    = (const float*)d_in[0];
  const int*   positions = (const int*)d_in[1];
  const float* wqkv      = (const float*)d_in[2];
  const float* wo        = (const float*)d_in[3];
  const float* q_norm_w  = (const float*)d_in[4];
  const float* k_norm_w  = (const float*)d_in[5];
  const float* ln1_w     = (const float*)d_in[6];
  const float* ln2_w     = (const float*)d_in[7];
  const float* gate_w    = (const float*)d_in[8];
  const float* w_gate    = (const float*)d_in[9];
  const float* w_up      = (const float*)d_in[10];
  const float* w_down    = (const float*)d_in[11];
  float* out = (float*)d_out;

  char* cur = (char*)d_ws;
  auto alloc = [&](size_t bytes) {
    void* r = (void*)cur;
    cur += (bytes + 255) & ~(size_t)255;
    return r;
  };

  float*  h      = (float*)alloc((size_t)T_ * H_ * 4);
  float*  qkv    = (float*)alloc((size_t)T_ * QKVN * 4);
  __bf16* Qh     = (__bf16*)alloc((size_t)NQ_ * T_ * D_ * 2);
  __bf16* Kh     = (__bf16*)alloc((size_t)NKV_ * T_ * D_ * 2);
  __bf16* Vh     = (__bf16*)alloc((size_t)NKV_ * T_ * D_ * 2);
  float*  attn   = (float*)alloc((size_t)T_ * AD_ * 4);
  float*  tmp    = (float*)alloc((size_t)T_ * H_ * 4);
  float*  x1     = (float*)alloc((size_t)T_ * H_ * 4);
  float*  h2     = (float*)alloc((size_t)T_ * H_ * 4);
  float*  logits = (float*)alloc((size_t)T_ * E_ * 4);
  float*  topk_w = (float*)alloc((size_t)T_ * K_ * 4);
  int*    topk_i = (int*)alloc((size_t)T_ * K_ * 4);
  int*    slot_t = (int*)alloc((size_t)E_ * CAP_ * 4);
  float*  slot_w = (float*)alloc((size_t)E_ * CAP_ * 4);
  __bf16* buf    = (__bf16*)alloc((size_t)E_ * CAP_ * H_ * 2);
  float*  gbuf   = (float*)alloc((size_t)E_ * CAP_ * I_ * 4);
  float*  ubuf   = (float*)alloc((size_t)E_ * CAP_ * I_ * 4);
  __bf16* act    = (__bf16*)alloc((size_t)E_ * CAP_ * I_ * 2);

  const dim3 blk256(256);
  const long long nTH = (long long)T_ * H_;

  // 1) h = rmsnorm(x, ln1)
  rmsnorm_kernel<<<T_, 256, 0, stream>>>(hidden, ln1_w, h, H_);

  // 2) qkv = h @ wqkv   (2048 x 5120 x 2048)
  gemm_wmma_kernel<float, false><<<dim3(QKVN / BN, T_ / BM, 1), blk256, 0, stream>>>(
      h, wqkv, qkv, T_, QKVN, H_, H_, QKVN, QKVN, 0, 0, 0, nullptr, nullptr);

  // 3) per-head rmsnorm + rope -> bf16 head-major Q/K/V
  rope_kernel<<<T_, 128, 0, stream>>>(qkv, positions, q_norm_w, k_norm_w, Qh, Kh, Vh);

  // 4) causal flash attention
  attn_kernel<<<dim3(T_ / 64, NQ_), 128, 0, stream>>>(Qh, Kh, Vh, attn);

  // 5) tmp = attn @ wo  (2048 x 2048 x 4096)
  gemm_wmma_kernel<float, false><<<dim3(H_ / BN, T_ / BM, 1), blk256, 0, stream>>>(
      attn, wo, tmp, T_, H_, AD_, AD_, H_, H_, 0, 0, 0, nullptr, nullptr);

  // 6) x1 = hidden + tmp
  add_kernel<<<(unsigned)((nTH + 255) / 256), blk256, 0, stream>>>(hidden, tmp, x1, nTH);

  // 7) h2 = rmsnorm(x1, ln2)
  rmsnorm_kernel<<<T_, 256, 0, stream>>>(x1, ln2_w, h2, H_);

  // 8) logits = h2 @ gate_w  (2048 x 64 x 2048)
  gemm_wmma_kernel<float, false><<<dim3(1, T_ / BM, 1), blk256, 0, stream>>>(
      h2, gate_w, logits, T_, E_, H_, H_, E_, E_, 0, 0, 0, nullptr, nullptr);

  // 9) softmax + top-8
  topk_kernel<<<T_, E_, 0, stream>>>(logits, topk_w, topk_i);

  // 10) sequential routing (capacity CAP)
  route_kernel<<<1, 1, 0, stream>>>(topk_i, topk_w, slot_t, slot_w);

  // 11) dispatch into (E, CAP, H) bf16 buffer
  scatter_kernel<<<E_ * CAP_, blk256, 0, stream>>>(slot_t, h2, buf);

  // 12) g = buf @ w_gate ; u = buf @ w_up   (batched: 64 x [512 x 768 x 2048])
  gemm_wmma_kernel<__bf16, false><<<dim3(I_ / BN, CAP_ / BM, E_), blk256, 0, stream>>>(
      buf, w_gate, gbuf, CAP_, I_, H_, H_, I_, I_,
      (long long)CAP_ * H_, (long long)H_ * I_, (long long)CAP_ * I_, nullptr, nullptr);
  gemm_wmma_kernel<__bf16, false><<<dim3(I_ / BN, CAP_ / BM, E_), blk256, 0, stream>>>(
      buf, w_up, ubuf, CAP_, I_, H_, H_, I_, I_,
      (long long)CAP_ * H_, (long long)H_ * I_, (long long)CAP_ * I_, nullptr, nullptr);

  // 13) act = silu(g) * u  -> bf16
  const long long nECI = (long long)E_ * CAP_ * I_;
  silu_mul_kernel<<<(unsigned)((nECI + 255) / 256), blk256, 0, stream>>>(gbuf, ubuf, act, nECI);

  // 14) out = x1 (MoE contributions accumulate on top)
  copy_kernel<<<(unsigned)((nTH + 255) / 256), blk256, 0, stream>>>(x1, out, nTH);

  // 15) down-proj with fused weighted scatter-add: out[t] += w * (act[e] @ w_down[e])
  gemm_wmma_kernel<__bf16, true><<<dim3(H_ / BN, CAP_ / BM, E_), blk256, 0, stream>>>(
      act, w_down, out, CAP_, H_, I_, I_, H_, H_,
      (long long)CAP_ * I_, (long long)I_ * H_, 0, slot_t, slot_w);
}
